// SparseBlockIndexer_7748121002258
// MI455X (gfx1250) — matrix operand order repaired and checked by reading, not compiled, not run on hardware
//
#include <hip/hip_runtime.h>
#include <hip/hip_bf16.h>

#define D_MODEL 2048
#define NUM_HEADS 16
#define IDIM 128
#define NBLK 4096
#define BATCH 32
#define KEEP 64
#define NEG_MAX (-3.402823466e+38f)

typedef __attribute__((ext_vector_type(16))) __bf16 v16bf;
typedef __attribute__((ext_vector_type(8)))  float  v8f;

#if __has_builtin(__builtin_amdgcn_tensor_load_to_lds) && \
    __has_builtin(__builtin_amdgcn_s_wait_tensorcnt)
#define HAVE_TDM 1
#else
#define HAVE_TDM 0
#endif

#if __has_builtin(__builtin_amdgcn_sched_barrier)
#define SCHED_BARRIER() __builtin_amdgcn_sched_barrier(0)
#else
#define SCHED_BARRIER()
#endif

// round-to-nearest-even f32 -> bf16 via integer ops
__device__ __forceinline__ unsigned short f2bf_bits(float f) {
    union { float f; unsigned u; } v; v.f = f;
    unsigned r = (v.u + 0x7FFFu + ((v.u >> 16) & 1u)) >> 16;
    return (unsigned short)r;
}

#if HAVE_TDM
typedef unsigned int tdm_v4u __attribute__((ext_vector_type(4)));
typedef int          tdm_v8i __attribute__((ext_vector_type(8)));
typedef int          tdm_v4i __attribute__((ext_vector_type(4)));

// TDM: async 2D tile load, W_k slice [128 rows x 64 bf16], row stride 2048.
// D# group0: count=1 | lds_addr | global_addr(57b) | type=2
// D# group1: data_size=1(2B); tensor_dim0=64; tensor_dim1=128;
//            tile_dim0=64; tile_dim1=128; tensor_dim0_stride=2048
__device__ __forceinline__ void tdm_load_wk_slice(unsigned lds_off,
                                                  const unsigned short* gsrc) {
    unsigned long long ga = (unsigned long long)(size_t)gsrc;
    tdm_v4u g0;
    g0.x = 1u;                                       // count=1, user descriptor
    g0.y = lds_off;                                  // lds_addr (bytes)
    g0.z = (unsigned)(ga & 0xFFFFFFFFu);             // global_addr lo
    g0.w = (unsigned)((ga >> 32) & 0x01FFFFFFu) | (2u << 30);  // addr hi | type=2
    tdm_v8i g1;
    g1[0] = (int)(1u << 16);                         // data_size = 2 bytes
    g1[1] = (int)(64u << 16);                        // tensor_dim0 = 64 (lo16)
    g1[2] = (int)(128u << 16);                       // tensor_dim1 = 128 (lo16)
    g1[3] = (int)(64u << 16);                        // tile_dim0 = 64
    g1[4] = (int)(128u);                             // tile_dim1 = 128, tile_dim2 = 0
    g1[5] = (int)(2048u);                            // tensor_dim0_stride lo32
    g1[6] = 0;
    g1[7] = 0;
    tdm_v4i gz = {0, 0, 0, 0};
#if defined(__clang_major__) && (__clang_major__ >= 23)
    tdm_v8i gz8 = {0, 0, 0, 0, 0, 0, 0, 0};
    __builtin_amdgcn_tensor_load_to_lds(g0, g1, gz, gz, gz8, 0);
#else
    __builtin_amdgcn_tensor_load_to_lds(g0, g1, gz, gz, 0);
#endif
}
#endif  // HAVE_TDM

// ---------------------------------------------------------------------------
// Kernel W: convert W_k [128 x 2048] f32 -> bf16 (stays in L2: 0.5 MB)
// ---------------------------------------------------------------------------
__global__ __launch_bounds__(256) void convert_wk_kernel(
    const float* __restrict__ wk, unsigned short* __restrict__ out) {
    int i = blockIdx.x * 256 + threadIdx.x;
    out[i] = f2bf_bits(wk[i]);
}

// ---------------------------------------------------------------------------
// Kernel A: per-batch query prep: q_lat RMSNorm, q_heads, head_w softmax
// ---------------------------------------------------------------------------
__global__ __launch_bounds__(128) void query_prep_kernel(
    const float* __restrict__ query,   // [B, D]
    const float* __restrict__ W_qd,    // [I, D]
    const float* __restrict__ b_qd,    // [I]
    const float* __restrict__ W_qu,    // [H*I, I]
    const float* __restrict__ b_qu,    // [H*I]
    const float* __restrict__ W_hw,    // [H, D]
    const float* __restrict__ b_hw,    // [H]
    const float* __restrict__ g_q,     // [I]
    float* __restrict__ q_heads,       // [B, H*I]
    float* __restrict__ head_w)        // [B, H]
{
    __shared__ float s_query[D_MODEL];
    __shared__ float s_lat[IDIM];
    __shared__ float s_red[128];
    __shared__ float s_hv[NUM_HEADS];

    const int b = blockIdx.x;
    const int tid = threadIdx.x;

    for (int i = tid; i < D_MODEL; i += 128)
        s_query[i] = query[(size_t)b * D_MODEL + i];
    __syncthreads();

    float x = b_qd[tid];
    {
        const float* wrow = W_qd + (size_t)tid * D_MODEL;
        for (int d = 0; d < D_MODEL; ++d) x += s_query[d] * wrow[d];
    }
    s_red[tid] = x * x;
    __syncthreads();
    for (int s = 64; s > 0; s >>= 1) {
        if (tid < s) s_red[tid] += s_red[tid + s];
        __syncthreads();
    }
    float rstd = rsqrtf(s_red[0] * (1.0f / IDIM) + 1e-6f);
    __syncthreads();
    s_lat[tid] = x * rstd * g_q[tid];

    if (tid < NUM_HEADS) {
        float hv = b_hw[tid];
        const float* wrow = W_hw + (size_t)tid * D_MODEL;
        for (int d = 0; d < D_MODEL; ++d) hv += s_query[d] * wrow[d];
        s_hv[tid] = hv;
    }
    __syncthreads();
    if (tid < NUM_HEADS) {
        float mx = NEG_MAX;
        for (int j = 0; j < NUM_HEADS; ++j) mx = fmaxf(mx, s_hv[j]);
        float sum = 0.f;
        for (int j = 0; j < NUM_HEADS; ++j) sum += expf(s_hv[j] - mx);
        head_w[b * NUM_HEADS + tid] = expf(s_hv[tid] - mx) / sum;
    }

    for (int j = tid; j < NUM_HEADS * IDIM; j += 128) {
        float acc = b_qu[j];
        const float* wrow = W_qu + (size_t)j * IDIM;
        #pragma unroll 8
        for (int i = 0; i < IDIM; ++i) acc += s_lat[i] * wrow[i];
        q_heads[(size_t)b * NUM_HEADS * IDIM + j] = acc;
    }
}

// ---------------------------------------------------------------------------
// Kernel B: WMMA key-projection + score kernel.
// Grid: B * (N/128) = 1024 workgroups x 256 threads (8 waves).
// W_k is staged 64 K-elements at a time into LDS by the Tensor Data Mover
// (double-buffered, overlapped with 16 WMMAs via TENSORcnt), shared by all
// 8 waves. B-fragment ds_loads are batched ahead of the WMMA group via a
// scheduler barrier so the matrix pipe runs back-to-back.
// ---------------------------------------------------------------------------
__global__ __launch_bounds__(256) void key_score_kernel(
    const float* __restrict__ blocks,          // [B, N, D] f32
    const unsigned short* __restrict__ wk_bf,  // [I, D] bf16 bits
    const float* __restrict__ b_k,             // [I]
    const float* __restrict__ g_k,             // [I]
    const float* __restrict__ q_heads,         // [B, H, I]
    const float* __restrict__ head_w,          // [B, H]
    const unsigned char* __restrict__ mask,    // [B, N] bool
    float* __restrict__ scores)                // [B, N]
{
    __shared__ float s_q[NUM_HEADS * IDIM];
    __shared__ float s_hw[NUM_HEADS];
    __shared__ float s_bk[IDIM];
    __shared__ float s_gk[IDIM];
#if HAVE_TDM
    __shared__ __align__(32) unsigned short s_wk[2][IDIM * 64];  // 2 x 16KB
#endif

    const int wg   = blockIdx.x;
    const int b    = wg >> 5;
    const int tile = wg & 31;
    const int tid  = threadIdx.x;

    for (int i = tid; i < NUM_HEADS * IDIM; i += 256)
        s_q[i] = q_heads[(size_t)b * NUM_HEADS * IDIM + i];
    if (tid < NUM_HEADS) s_hw[tid] = head_w[b * NUM_HEADS + tid];
    if (tid < IDIM) { s_bk[tid] = b_k[tid]; s_gk[tid] = g_k[tid]; }

    const int wave = tid >> 5;
    const int lane = tid & 31;
    const int m  = lane & 15;
    const int hi = lane >> 4;

    const int n0 = tile * 128 + wave * 16;
    const float* rowp = blocks + ((size_t)b * NBLK + n0 + m) * D_MODEL;

#if HAVE_TDM
    // scalarized wave id so non-issuing waves *branch around* the TDM op
    const int wv = __builtin_amdgcn_readfirstlane((int)threadIdx.x) >> 5;
    if (wv == 0) {
        tdm_load_wk_slice((unsigned)(size_t)&s_wk[0][0], wk_bf);
        __builtin_amdgcn_s_wait_tensorcnt(0);
    }
#endif
    __syncthreads();

    v8f c[8];
    #pragma unroll
    for (int t = 0; t < 8; ++t) c[t] = (v8f){0.f,0.f,0.f,0.f,0.f,0.f,0.f,0.f};

    // preload raw A for kb = 0
    float4 ar0 = *(const float4*)(rowp + 16 * hi);
    float4 ar1 = *(const float4*)(rowp + 16 * hi + 4);
    float4 ar2 = *(const float4*)(rowp + 16 * hi + 8);
    float4 ar3 = *(const float4*)(rowp + 16 * hi + 12);

    for (int kb2 = 0; kb2 < 32; ++kb2) {
#if HAVE_TDM
        // kick DMA of next 64-wide W_k slice; overlaps with 16 WMMAs below
        if (kb2 < 31 && wv == 0)
            tdm_load_wk_slice((unsigned)(size_t)&s_wk[(kb2 + 1) & 1][0],
                              wk_bf + (size_t)(kb2 + 1) * 64);
#endif
        #pragma unroll
        for (int j = 0; j < 2; ++j) {
            const int kb = kb2 * 2 + j;
            const int k = kb * 32;

            if (kb < 60) __builtin_prefetch(rowp + k + 128, 0, 1);

            // prefetch raw A for next 32-K step into registers
            float4 an0 = ar0, an1 = ar1, an2 = ar2, an3 = ar3;
            if (kb < 63) {
                const float* np = rowp + k + 32;
                an0 = *(const float4*)(np + 16 * hi);
                an1 = *(const float4*)(np + 16 * hi + 4);
                an2 = *(const float4*)(np + 16 * hi + 8);
                an3 = *(const float4*)(np + 16 * hi + 12);
            }

            // batch all 16 B-fragment ds_loads first
            v16bf bfr[8];
#if HAVE_TDM
            {
                const unsigned short* wkp = &s_wk[kb2 & 1][j * 32 + 16 * hi];
                #pragma unroll
                for (int t = 0; t < 8; ++t)
                    bfr[t] = *(const v16bf*)(wkp + (t * 16 + m) * 64);
            }
#else
            #pragma unroll
            for (int t = 0; t < 8; ++t)
                bfr[t] = *(const v16bf*)(wk_bf + (size_t)(t * 16 + m) * D_MODEL +
                                         k + 16 * hi);
#endif

            // A fragment (contiguous K order, matching B): e -> K=k+16*hi+e
            // conversion VALU overlaps the LDS latency of the batched loads
            unsigned short abits[16];
            abits[0]=f2bf_bits(ar0.x); abits[1]=f2bf_bits(ar0.y);
            abits[2]=f2bf_bits(ar0.z); abits[3]=f2bf_bits(ar0.w);
            abits[4]=f2bf_bits(ar1.x); abits[5]=f2bf_bits(ar1.y);
            abits[6]=f2bf_bits(ar1.z); abits[7]=f2bf_bits(ar1.w);
            abits[8]=f2bf_bits(ar2.x); abits[9]=f2bf_bits(ar2.y);
            abits[10]=f2bf_bits(ar2.z); abits[11]=f2bf_bits(ar2.w);
            abits[12]=f2bf_bits(ar3.x); abits[13]=f2bf_bits(ar3.y);
            abits[14]=f2bf_bits(ar3.z); abits[15]=f2bf_bits(ar3.w);
            v16bf a;
            __builtin_memcpy(&a, abits, sizeof(a));

            // keep loads above, WMMAs below: one ds wait, 8 WMMAs back-to-back
            SCHED_BARRIER();

            #pragma unroll
            for (int t = 0; t < 8; ++t)
                c[t] = __builtin_amdgcn_wmma_f32_16x16x32_bf16(
                    false, a, false, bfr[t], (short)0, c[t], false, false);

            ar0 = an0; ar1 = an1; ar2 = an2; ar3 = an3;
        }
#if HAVE_TDM
        if (wv == 0) __builtin_amdgcn_s_wait_tensorcnt(0);
        __syncthreads();   // next slice ready; all waves done with old buffer
#endif
    }

    // ---- epilogue: +b_k, RMSNorm rows, per-head relu-dot, head-weighted sum
    float rstd[8];
    #pragma unroll
    for (int r = 0; r < 8; ++r) {
        float acc = 0.f;
        #pragma unroll
        for (int t = 0; t < 8; ++t) {
            float x = c[t][r] + s_bk[t * 16 + m];
            c[t][r] = x;
            acc += x * x;
        }
        #pragma unroll
        for (int s = 1; s < 16; s <<= 1) acc += __shfl_xor(acc, s, 32);
        rstd[r] = rsqrtf(acc * (1.0f / IDIM) + 1e-6f);
    }
    #pragma unroll
    for (int t = 0; t < 8; ++t) {
        const float gk = s_gk[t * 16 + m];
        #pragma unroll
        for (int r = 0; r < 8; ++r) c[t][r] = c[t][r] * rstd[r] * gk;
    }

    float rowsc[8] = {0.f,0.f,0.f,0.f,0.f,0.f,0.f,0.f};
    for (int h = 0; h < NUM_HEADS; ++h) {
        float qv[8];
        #pragma unroll
        for (int t = 0; t < 8; ++t) qv[t] = s_q[h * IDIM + t * 16 + m];
        const float w = s_hw[h];
        #pragma unroll
        for (int r = 0; r < 8; ++r) {
            float p = 0.f;
            #pragma unroll
            for (int t = 0; t < 8; ++t) p += qv[t] * c[t][r];
            #pragma unroll
            for (int s = 1; s < 16; s <<= 1) p += __shfl_xor(p, s, 32);
            rowsc[r] += w * fmaxf(p, 0.f);
        }
    }

    if (m == 0) {
        #pragma unroll
        for (int r = 0; r < 8; ++r) {
            const int n = n0 + 8 * hi + r;
            float sc = rowsc[r];
            if (mask[(size_t)b * NBLK + n]) sc = NEG_MAX;
            scores[(size_t)b * NBLK + n] = sc;
        }
    }
}

// ---------------------------------------------------------------------------
// Kernel C: per-batch iterative top-64 (scores LDS-resident)
// ---------------------------------------------------------------------------
__global__ __launch_bounds__(256) void topk_kernel(
    const float* __restrict__ scores, int* __restrict__ idx_out) {
    __shared__ float s[NBLK];
    __shared__ float rv[256];
    __shared__ int   ri[256];

    const int b = blockIdx.x;
    const int tid = threadIdx.x;
    for (int i = tid; i < NBLK; i += 256) s[i] = scores[(size_t)b * NBLK + i];
    __syncthreads();

    for (int t = 0; t < KEEP; ++t) {
        float best = -__builtin_inff();
        int   bi   = 0x7fffffff;
        for (int i = tid; i < NBLK; i += 256) {
            float v = s[i];
            if (v > best || (v == best && i < bi)) { best = v; bi = i; }
        }
        rv[tid] = best; ri[tid] = bi;
        __syncthreads();
        for (int step = 128; step > 0; step >>= 1) {
            if (tid < step) {
                float ov = rv[tid + step]; int oi = ri[tid + step];
                if (ov > rv[tid] || (ov == rv[tid] && oi < ri[tid])) {
                    rv[tid] = ov; ri[tid] = oi;
                }
            }
            __syncthreads();
        }
        if (tid == 0) {
            int win = ri[0];
            idx_out[b * KEEP + t] = win;
            s[win] = -__builtin_inff();
        }
        __syncthreads();
    }
}

// ---------------------------------------------------------------------------
// Kernel D: gather selected blocks + mask flags into d_out
// ---------------------------------------------------------------------------
__global__ __launch_bounds__(256) void gather_kernel(
    const float* __restrict__ blocks,
    const unsigned char* __restrict__ mask,
    const int* __restrict__ idx,
    float* __restrict__ out) {
    const int row = blockIdx.x;
    const int b = row >> 6;
    const int n = idx[row];
    const float* src = blocks + ((size_t)b * NBLK + n) * D_MODEL;
    float* dst = out + (size_t)row * D_MODEL;
    for (int i = threadIdx.x; i < D_MODEL; i += 256) dst[i] = src[i];
    if (threadIdx.x == 0)
        out[(size_t)BATCH * KEEP * D_MODEL + row] =
            mask[(size_t)b * NBLK + n] ? 1.0f : 0.0f;
}

// ---------------------------------------------------------------------------
extern "C" void kernel_launch(void* const* d_in, const int* in_sizes, int n_in,
                              void* d_out, int out_size, void* d_ws, size_t ws_size,
                              hipStream_t stream) {
    const float*         query  = (const float*)d_in[0];
    const float*         blocks = (const float*)d_in[1];
    const unsigned char* bmask  = (const unsigned char*)d_in[2];
    // d_in[3] = top_k (device scalar, == 64; KEEP hardcoded)
    const float* W_qd = (const float*)d_in[4];
    const float* b_qd = (const float*)d_in[5];
    const float* W_qu = (const float*)d_in[6];
    const float* b_qu = (const float*)d_in[7];
    const float* W_k  = (const float*)d_in[8];
    const float* b_k  = (const float*)d_in[9];
    const float* W_hw = (const float*)d_in[10];
    const float* b_hw = (const float*)d_in[11];
    const float* g_q  = (const float*)d_in[12];
    const float* g_k  = (const float*)d_in[13];

    char* ws = (char*)d_ws;
    unsigned short* wk_bf   = (unsigned short*)(ws);            // 512 KB
    float*          q_heads = (float*)(ws + 524288);            // 256 KB
    float*          head_w  = (float*)(ws + 786432);            //   2 KB
    float*          scores  = (float*)(ws + 788480);            // 512 KB
    int*            idxb    = (int*)(ws + 1312768);             //   8 KB

    convert_wk_kernel<<<(IDIM * D_MODEL) / 256, 256, 0, stream>>>(W_k, wk_bf);
    query_prep_kernel<<<BATCH, 128, 0, stream>>>(
        query, W_qd, b_qd, W_qu, b_qu, W_hw, b_hw, g_q, q_heads, head_w);
    key_score_kernel<<<BATCH * (NBLK / 128), 256, 0, stream>>>(
        blocks, wk_bf, b_k, g_k, q_heads, head_w, bmask, scores);
    topk_kernel<<<BATCH, 256, 0, stream>>>(scores, idxb);
    gather_kernel<<<BATCH * KEEP, 256, 0, stream>>>(blocks, bmask, idxb,
                                                    (float*)d_out);
}
